// MoleculeNet_30167850287488
// MI455X (gfx1250) — matrix operand order repaired and checked by reading, not compiled, not run on hardware
//
#include <hip/hip_runtime.h>

typedef __attribute__((ext_vector_type(16))) __bf16 v16bf;
typedef __attribute__((ext_vector_type(8)))  float  v8f;

#define TB 256

__device__ __forceinline__ void bf16_split(float v, __bf16& h, __bf16& l) {
  h = (__bf16)v;
  l = (__bf16)(v - (float)h);
}

// ---------------- degree / normalization ----------------
__global__ void k_deg_init(float* deg, int n) {
  int i = blockIdx.x * blockDim.x + threadIdx.x;
  if (i < n) deg[i] = 1.0f;                       // self loop
}

__global__ void k_deg_acc(const int* __restrict__ dst, float* deg, int nE) {
  int e = blockIdx.x * blockDim.x + threadIdx.x;
  if (e < nE) atomicAdd(&deg[dst[e]], 1.0f);
}

__global__ void k_deg_rsqrt(float* deg, int n) {
  int i = blockIdx.x * blockDim.x + threadIdx.x;
  if (i < n) deg[i] = rsqrtf(deg[i]);
}

// ---------------- one-time weight fragment precompute (split-bf16, B layout) ----------------
// W1 [64,16]: frag1[(kb2*2 + {0=hi,1=lo})*32 + lane]
__global__ void k_wsplit1(const float* __restrict__ W1, v16bf* __restrict__ frag) {
  int lane = threadIdx.x;
  if (lane >= 32) return;
  int m = lane & 15, hi = lane >> 4;
  for (int kb2 = 0; kb2 < 2; ++kb2) {
    v16bf bh, bl;
    #pragma unroll
    for (int i = 0; i < 16; ++i) {
      __bf16 th, tl; bf16_split(W1[(kb2 * 32 + hi * 8 + i + (i >= 8 ? 8 : 0)) * 16 + m], th, tl);
      bh[i] = th; bl[i] = tl;
    }
    frag[(kb2 * 2 + 0) * 32 + lane] = bh;
    frag[(kb2 * 2 + 1) * 32 + lane] = bl;
  }
}

// W2 [16,16], K 16..31 zero-padded: frag2[{0=hi,1=lo}*32 + lane]
__global__ void k_wsplit2(const float* __restrict__ W2, v16bf* __restrict__ frag) {
  int lane = threadIdx.x;
  if (lane >= 32) return;
  int m = lane & 15, hi = lane >> 4;
  v16bf bh, bl;
  #pragma unroll
  for (int i = 0; i < 16; ++i) {
    float wv = (i >= 8) ? 0.0f : W2[(hi * 8 + i) * 16 + m];
    __bf16 th, tl; bf16_split(wv, th, tl);
    bh[i] = th; bl[i] = tl;
  }
  frag[0 * 32 + lane] = bh;
  frag[1 * 32 + lane] = bl;
}

// ---------------- GEMM1: h = x[nN,64] @ W1[64,16], split-bf16 WMMA ----------------
__global__ __launch_bounds__(TB) void k_gemm1(const float* __restrict__ x,
                                              const v16bf* __restrict__ wfrag,
                                              float* __restrict__ h, int nN) {
  const int wave = threadIdx.x >> 5;
  const int lane = threadIdx.x & 31;
  const long long base = ((long long)blockIdx.x * 8 + wave) * 16;
  if (base >= nN) return;                         // wave-uniform exit

  const int m    = lane & 15;                     // row in tile / col of B
  const int hi   = lane >> 4;                     // lane half
  const int koff = hi * 8;
  long long row = base + m;
  if (row >= nN) row = nN - 1;                    // clamp load (EXEC stays full)
  const float* xr = x + row * 64;

  v8f c = {};
  #pragma unroll
  for (int kb2 = 0; kb2 < 2; ++kb2) {
    const int kb = kb2 * 32;
    // A tile: two 32B chunks per lane -> b128 loads
    const float4 q0 = *(const float4*)(xr + kb + koff);
    const float4 q1 = *(const float4*)(xr + kb + koff + 4);
    const float4 q2 = *(const float4*)(xr + kb + 16 + koff);
    const float4 q3 = *(const float4*)(xr + kb + 16 + koff + 4);
    float av[16] = {q0.x,q0.y,q0.z,q0.w, q1.x,q1.y,q1.z,q1.w,
                    q2.x,q2.y,q2.z,q2.w, q3.x,q3.y,q3.z,q3.w};
    v16bf ah, al;
    #pragma unroll
    for (int i = 0; i < 16; ++i) {
      __bf16 th, tl; bf16_split(av[i], th, tl);
      ah[i] = th; al[i] = tl;
    }
    const v16bf bh = wfrag[(kb2 * 2 + 0) * 32 + lane];   // precomputed, L2-resident
    const v16bf bl = wfrag[(kb2 * 2 + 1) * 32 + lane];
    c = __builtin_amdgcn_wmma_f32_16x16x32_bf16(false, ah, false, bh, (short)0, c, false, false);
    c = __builtin_amdgcn_wmma_f32_16x16x32_bf16(false, ah, false, bl, (short)0, c, false, false);
    c = __builtin_amdgcn_wmma_f32_16x16x32_bf16(false, al, false, bh, (short)0, c, false, false);
  }

  if (base + 16 <= (long long)nN) {               // full tile: unguarded stores
    float* hb = h + (base + hi * 8) * 16 + m;
    #pragma unroll
    for (int v = 0; v < 8; ++v) hb[v * 16] = c[v];
  } else {
    #pragma unroll
    for (int v = 0; v < 8; ++v) {
      long long r = base + v + hi * 8;
      if (r < nN) h[r * 16 + m] = c[v];
    }
  }
}

// ---------------- GEMM2: h = relu(in)[nN,16] @ W2[16,16], split-bf16, K padded ----------------
__global__ __launch_bounds__(TB) void k_gemm2(const float* __restrict__ hin,
                                              const v16bf* __restrict__ wfrag,
                                              float* __restrict__ hout, int nN) {
  const int wave = threadIdx.x >> 5;
  const int lane = threadIdx.x & 31;
  const long long base = ((long long)blockIdx.x * 8 + wave) * 16;
  if (base >= nN) return;

  const int m    = lane & 15;
  const int hi   = lane >> 4;
  const int koff = hi * 8;
  long long row = base + m;
  if (row >= nN) row = nN - 1;
  const float* hr = hin + row * 16;

  const float4 q0 = *(const float4*)(hr + koff);
  const float4 q1 = *(const float4*)(hr + koff + 4);
  float av[8] = {q0.x,q0.y,q0.z,q0.w, q1.x,q1.y,q1.z,q1.w};

  v16bf ah, al;
  #pragma unroll
  for (int i = 0; i < 8; ++i) {
    __bf16 th, tl; bf16_split(fmaxf(av[i], 0.0f), th, tl);   // relu fused
    ah[i] = th;               al[i] = tl;
    ah[i + 8] = (__bf16)0.0f; al[i + 8] = (__bf16)0.0f;      // K = 16..31 padding
  }
  const v16bf bh = wfrag[0 * 32 + lane];
  const v16bf bl = wfrag[1 * 32 + lane];

  v8f c = {};
  c = __builtin_amdgcn_wmma_f32_16x16x32_bf16(false, ah, false, bh, (short)0, c, false, false);
  c = __builtin_amdgcn_wmma_f32_16x16x32_bf16(false, ah, false, bl, (short)0, c, false, false);
  c = __builtin_amdgcn_wmma_f32_16x16x32_bf16(false, al, false, bh, (short)0, c, false, false);

  if (base + 16 <= (long long)nN) {
    float* hb = hout + (base + hi * 8) * 16 + m;
    #pragma unroll
    for (int v = 0; v < 8; ++v) hb[v * 16] = c[v];
  } else {
    #pragma unroll
    for (int v = 0; v < 8; ++v) {
      long long r = base + v + hi * 8;
      if (r < nN) hout[r * 16 + m] = c[v];
    }
  }
}

// ---------------- accumulator init: self-loop message + bias (float4) ----------------
__global__ void k_init_accum(const float4* __restrict__ h, const float* __restrict__ dinv,
                             const float4* __restrict__ bias4, float4* __restrict__ out, int nN) {
  long long gid = (long long)blockIdx.x * blockDim.x + threadIdx.x;
  if (gid >= (long long)nN * 4) return;
  int i = (int)(gid >> 2), c4 = (int)(gid & 3);
  float di = dinv[i], s = di * di;
  float4 hv = h[gid], bv = bias4[c4], o;
  o.x = hv.x * s + bv.x;  o.y = hv.y * s + bv.y;
  o.z = hv.z * s + bv.z;  o.w = hv.w * s + bv.w;
  out[gid] = o;
}

// ---------------- edge scatter: out[d] += h[s] * dinv[s]*dinv[d] ----------------
__global__ void k_scatter(const int* __restrict__ src, const int* __restrict__ dst,
                          const float* __restrict__ dinv, const float* __restrict__ h,
                          float* out, int nE) {
  long long gid = (long long)blockIdx.x * blockDim.x + threadIdx.x;
  if (gid >= (long long)nE * 16) return;
  int e = (int)(gid >> 4), c = (int)(gid & 15);
  int s = src[e], d = dst[e];
  float norm = dinv[s] * dinv[d];
  atomicAdd(&out[(long long)d * 16 + c], h[(long long)s * 16 + c] * norm);
}

// ---------------- pooling ----------------
__global__ void k_zero(float* p, long long n) {
  long long gid = (long long)blockIdx.x * blockDim.x + threadIdx.x;
  if (gid < n) p[gid] = 0.0f;
}

__global__ void k_pool(const int* __restrict__ batch, const float* __restrict__ h,
                       float* pooled, int nN) {
  long long gid = (long long)blockIdx.x * blockDim.x + threadIdx.x;
  if (gid >= (long long)nN * 16) return;
  int i = (int)(gid >> 4), c = (int)(gid & 15);
  atomicAdd(&pooled[(long long)batch[i] * 16 + c], h[gid]);
}

// ---------------- MLP head: relu -> [16x100] relu -> [100x27] ----------------
__global__ __launch_bounds__(TB) void k_head(const float* __restrict__ pooled,
                                             const float* __restrict__ LW1,
                                             const float* __restrict__ Lb1,
                                             const float* __restrict__ LW2,
                                             const float* __restrict__ Lb2,
                                             float* __restrict__ out, int nG) {
  int g = blockIdx.x * blockDim.x + threadIdx.x;
  if (g >= nG) return;
  float p[16];
  #pragma unroll
  for (int k = 0; k < 16; ++k) p[k] = fmaxf(pooled[(long long)g * 16 + k], 0.0f);
  float acc[27];
  #pragma unroll
  for (int c = 0; c < 27; ++c) acc[c] = Lb2[c];
  for (int j = 0; j < 100; ++j) {
    float t = Lb1[j];
    #pragma unroll
    for (int k = 0; k < 16; ++k) t += p[k] * LW1[k * 100 + j];
    t = fmaxf(t, 0.0f);
    #pragma unroll
    for (int c = 0; c < 27; ++c) acc[c] += t * LW2[j * 27 + c];
  }
  #pragma unroll
  for (int c = 0; c < 27; ++c) out[(long long)g * 27 + c] = acc[c];
}

static inline int cdivll(long long a, long long b) { return (int)((a + b - 1) / b); }

extern "C" void kernel_launch(void* const* d_in, const int* in_sizes, int n_in,
                              void* d_out, int out_size, void* d_ws, size_t ws_size,
                              hipStream_t stream) {
  const float* x   = (const float*)d_in[0];
  const int*   ei  = (const int*)d_in[1];
  const int*   bat = (const int*)d_in[2];
  const float* W1  = (const float*)d_in[3];
  const float* b1  = (const float*)d_in[4];
  const float* W2  = (const float*)d_in[5];
  const float* b2  = (const float*)d_in[6];
  const float* LW1 = (const float*)d_in[7];
  const float* Lb1 = (const float*)d_in[8];
  const float* LW2 = (const float*)d_in[9];
  const float* Lb2 = (const float*)d_in[10];

  const int nN = in_sizes[0] / 64;
  const int nE = in_sizes[1] / 2;
  const int nG = out_size / 27;
  const int* src = ei;
  const int* dst = ei + nE;

  // workspace layout: weight fragments first (32B aligned), then float arrays
  v16bf* wfrag1 = (v16bf*)d_ws;                        // 128 v16bf = 4KB
  v16bf* wfrag2 = wfrag1 + 128;                        //  64 v16bf = 2KB
  float* dinv   = (float*)(wfrag2 + 64);               // nN
  float* hraw   = dinv + nN;                           // nN*16
  float* accum  = hraw + (size_t)nN * 16;              // nN*16
  float* pooled = accum + (size_t)nN * 16;             // nG*16

  const long long nodeCh  = (long long)nN * 16;
  const long long nodeCh4 = (long long)nN * 4;
  const long long edgeCh  = (long long)nE * 16;
  const int gTiles = cdivll(nN, 128);                  // 8 waves/block, 16 rows/wave

  // normalization + weight fragment precompute
  k_deg_init <<<cdivll(nN, TB), TB, 0, stream>>>(dinv, nN);
  k_deg_acc  <<<cdivll(nE, TB), TB, 0, stream>>>(dst, dinv, nE);
  k_deg_rsqrt<<<cdivll(nN, TB), TB, 0, stream>>>(dinv, nN);
  k_wsplit1  <<<1, 32, 0, stream>>>(W1, wfrag1);
  k_wsplit2  <<<1, 32, 0, stream>>>(W2, wfrag2);

  // layer 1
  k_gemm1     <<<gTiles, TB, 0, stream>>>(x, wfrag1, hraw, nN);
  k_init_accum<<<cdivll(nodeCh4, TB), TB, 0, stream>>>((const float4*)hraw, dinv,
                                                       (const float4*)b1, (float4*)accum, nN);
  k_scatter   <<<cdivll(edgeCh, TB), TB, 0, stream>>>(src, dst, dinv, hraw, accum, nE);

  // layer 2 (relu fused into gemm2 input)
  k_gemm2     <<<gTiles, TB, 0, stream>>>(accum, wfrag2, hraw, nN);
  k_init_accum<<<cdivll(nodeCh4, TB), TB, 0, stream>>>((const float4*)hraw, dinv,
                                                       (const float4*)b2, (float4*)accum, nN);
  k_scatter   <<<cdivll(edgeCh, TB), TB, 0, stream>>>(src, dst, dinv, hraw, accum, nE);

  // pooling + head
  k_zero<<<cdivll((long long)nG * 16, TB), TB, 0, stream>>>(pooled, (long long)nG * 16);
  k_pool<<<cdivll(nodeCh, TB), TB, 0, stream>>>(bat, accum, pooled, nN);
  k_head<<<cdivll(nG, TB), TB, 0, stream>>>(pooled, LW1, Lb1, LW2, Lb2, (float*)d_out, nG);
}
